// DeepSeekFeedForward_71451075936568
// MI455X (gfx1250) — matrix operand order repaired and checked by reading, not compiled, not run on hardware
//
#include <hip/hip_runtime.h>

// dims: B=4, T=1024, C=1024, H=1024, E=8, K=2 ; S = 4096
#define S_TOK 4096
#define CDIM  1024
#define HDIM  1024
#define NEXP  8
#define LDK   64   // K elements staged per barrier round (2 WMMA k-chunks)
#define STR   72   // LDS row stride in bf16 elements (144B: 16B-aligned, bank-padded)

typedef __attribute__((ext_vector_type(16))) __bf16 v16bf;
typedef __attribute__((ext_vector_type(8)))  float  v8f;
typedef __attribute__((ext_vector_type(8)))  unsigned short ushort8;

union FragBF { v16bf v; ushort8 h[2]; };

__device__ __forceinline__ unsigned short f2bf(float f) {
  union { float f; unsigned int u; } x; x.f = f;
  unsigned int r = x.u + 0x7FFFu + ((x.u >> 16) & 1u);  // round-nearest-even
  return (unsigned short)(r >> 16);
}

__device__ __forceinline__ v8f wmma_bf16(v16bf a, v16bf b, v8f c) {
  // D = A(16x32 bf16) x B(32x16 bf16) + C(16x16 f32)
  return __builtin_amdgcn_wmma_f32_16x16x32_bf16(
      false, a, false, b, (short)0, c, false, false);
}

// A fragment for k-chunk kc (K=32): lane L holds row (L&15).
// Lanes 0-15: K kc*32+{0..7,16..23}; lanes 16-31: +8 (ISA 7.12.2, 16-bit A 16x32)
__device__ __forceinline__ v16bf load_a_frag(const unsigned short* sA, int rowBase,
                                             int kc, int lane) {
  int r  = rowBase + (lane & 15);
  int kb = kc * 32 + ((lane < 16) ? 0 : 8);
  FragBF f;
  f.h[0] = *(const ushort8*)(sA + r * STR + kb);
  f.h[1] = *(const ushort8*)(sA + r * STR + kb + 16);
  return f.v;
}

// B fragment (LDS tile stored transposed as [col][k]):
// lanes 0-15 hold K=kc*32+{0..7,16..23} of column (L&15); lanes 16-31 hold +8
__device__ __forceinline__ v16bf load_b_frag(const unsigned short* sB, int colBase,
                                             int kc, int lane) {
  int c  = colBase + (lane & 15);
  int ko = kc * 32 + ((lane < 16) ? 0 : 8);
  FragBF f;
  f.h[0] = *(const ushort8*)(sB + c * STR + ko);
  f.h[1] = *(const ushort8*)(sB + c * STR + ko + 16);
  return f.v;
}

// Stage A tile: 128 rows x 64 k, row-major bf16 global -> LDS (coalesced 16B)
__device__ __forceinline__ void stage_a(unsigned short* sA, const unsigned short* A,
                                        int m0, int k0, int lda, int tid) {
#pragma unroll
  for (int i = 0; i < 4; ++i) {
    int chunk = tid + i * 256;      // 1024 chunks of 8 halfs
    int row = chunk >> 3;
    int c8  = (chunk & 7) * 8;
    *(ushort8*)(sA + row * STR + c8) =
        *(const ushort8*)(A + (size_t)(m0 + row) * lda + k0 + c8);
  }
}

// Stage B tile 64(k) x 64(n) from row-major [K,N] global, transposed to [n][k] in LDS.
// Each thread handles a K-row pair of 8 columns; pairs packed -> 8x ds_store_b32.
__device__ __forceinline__ void stage_b(unsigned short* sB, const unsigned short* W,
                                        int k0, int n0, int ldb, int tid) {
  int k = (tid >> 3) * 2;    // 0,2,..,62
  int n = (tid & 7) * 8;     // 0..56
  ushort8 r0 = *(const ushort8*)(W + (size_t)(k0 + k)     * ldb + n0 + n);
  ushort8 r1 = *(const ushort8*)(W + (size_t)(k0 + k + 1) * ldb + n0 + n);
#pragma unroll
  for (int j = 0; j < 8; ++j) {
    unsigned int p = (unsigned int)r0[j] | ((unsigned int)r1[j] << 16);
    *(unsigned int*)(sB + (n + j) * STR + k) = p;   // k even -> 4B aligned
  }
}

// ---------------- cast f32 -> bf16 (4 elements per thread) ----------------
__global__ void cast_bf16_kernel(const float* __restrict__ src,
                                 unsigned short* __restrict__ dst, int n4) {
  int i = blockIdx.x * blockDim.x + threadIdx.x;
  if (i < n4) {
    float4 v = ((const float4*)src)[i];
    ushort4 o;
    o.x = f2bf(v.x); o.y = f2bf(v.y); o.z = f2bf(v.z); o.w = f2bf(v.w);
    ((ushort4*)dst)[i] = o;
  }
}

// ---------------- router: softmax + top-2 -> dense dispatch weights -------
__global__ __launch_bounds__(256) void gate_kernel(const float* __restrict__ x,
                                                   const float* __restrict__ wg,
                                                   const float* __restrict__ bias,
                                                   float* __restrict__ wdense) {
  int token = blockIdx.x * 8 + (threadIdx.x >> 5);   // one wave32 per token
  int lane  = threadIdx.x & 31;
  const float* xr = x + (size_t)token * CDIM;
  float acc[NEXP];
#pragma unroll
  for (int e = 0; e < NEXP; ++e) acc[e] = 0.0f;
  for (int c = lane; c < CDIM; c += 32) {
    float xv = xr[c];
    const float* wr = wg + (size_t)c * NEXP;
#pragma unroll
    for (int e = 0; e < NEXP; ++e) acc[e] += xv * wr[e];
  }
#pragma unroll
  for (int e = 0; e < NEXP; ++e)
#pragma unroll
    for (int off = 16; off > 0; off >>= 1)
      acc[e] += __shfl_xor(acc[e], off, 32);
  if (lane == 0) {
    float lg[NEXP], mx = -1e30f;
#pragma unroll
    for (int e = 0; e < NEXP; ++e) { lg[e] = acc[e] + bias[e]; mx = fmaxf(mx, lg[e]); }
    float den = 0.0f;
#pragma unroll
    for (int e = 0; e < NEXP; ++e) { lg[e] = __expf(lg[e] - mx); den += lg[e]; }
    float inv = 1.0f / den;
#pragma unroll
    for (int e = 0; e < NEXP; ++e) lg[e] *= inv;     // probs
    int i0 = 0;
#pragma unroll
    for (int e = 1; e < NEXP; ++e) if (lg[e] > lg[i0]) i0 = e;
    int i1 = (i0 == 0) ? 1 : 0;
#pragma unroll
    for (int e = 0; e < NEXP; ++e) if (e != i0 && lg[e] > lg[i1]) i1 = e;
    float* row = wdense + (size_t)token * NEXP;
#pragma unroll
    for (int e = 0; e < NEXP; ++e)
      row[e] = (e == i0) ? lg[i0] : ((e == i1) ? lg[i1] : 0.0f);
  }
}

// ---------------- fused SwiGLU GEMM: Hid = silu(X@W[:, :H]) * (X@W[:, H:]) -
__global__ __launch_bounds__(256) void gemm_swiglu_kernel(
    const unsigned short* __restrict__ X,   // [S, C] bf16
    const unsigned short* __restrict__ W,   // [C, 2H] bf16
    unsigned short* __restrict__ Hid) {     // [S, H] bf16
  __shared__ unsigned short sA [128 * STR];
  __shared__ unsigned short sBa[64 * STR];
  __shared__ unsigned short sBb[64 * STR];
  int tid = threadIdx.x, lane = tid & 31, wave = tid >> 5;
  int wm = wave & 3, wn = wave >> 2;            // 4x2 wave grid -> 128x64 tile
  int m0 = blockIdx.y * 128, n0 = blockIdx.x * 64;
  v8f accA[2][2] = {}; v8f accB[2][2] = {};
  for (int k0 = 0; k0 < CDIM; k0 += LDK) {
    stage_a(sA, X, m0, k0, CDIM, tid);
    stage_b(sBa, W, k0, n0,        2 * HDIM, tid);
    stage_b(sBb, W, k0, n0 + HDIM, 2 * HDIM, tid);
    __syncthreads();
#pragma unroll
    for (int kc = 0; kc < 2; ++kc) {
      v16bf a0  = load_a_frag(sA,  wm * 32,      kc, lane);
      v16bf a1  = load_a_frag(sA,  wm * 32 + 16, kc, lane);
      v16bf ba0 = load_b_frag(sBa, wn * 32,      kc, lane);
      v16bf ba1 = load_b_frag(sBa, wn * 32 + 16, kc, lane);
      v16bf bb0 = load_b_frag(sBb, wn * 32,      kc, lane);
      v16bf bb1 = load_b_frag(sBb, wn * 32 + 16, kc, lane);
      accA[0][0] = wmma_bf16(a0, ba0, accA[0][0]);
      accA[0][1] = wmma_bf16(a0, ba1, accA[0][1]);
      accA[1][0] = wmma_bf16(a1, ba0, accA[1][0]);
      accA[1][1] = wmma_bf16(a1, ba1, accA[1][1]);
      accB[0][0] = wmma_bf16(a0, bb0, accB[0][0]);
      accB[0][1] = wmma_bf16(a0, bb1, accB[0][1]);
      accB[1][0] = wmma_bf16(a1, bb0, accB[1][0]);
      accB[1][1] = wmma_bf16(a1, bb1, accB[1][1]);
    }
    __syncthreads();
  }
  int rofs = (lane < 16) ? 0 : 8;
  int cofs = lane & 15;
#pragma unroll
  for (int mt = 0; mt < 2; ++mt)
#pragma unroll
    for (int nt = 0; nt < 2; ++nt)
#pragma unroll
      for (int v = 0; v < 8; ++v) {
        int row = m0 + wm * 32 + mt * 16 + rofs + v;
        int col = n0 + wn * 32 + nt * 16 + cofs;
        float a = accA[mt][nt][v], b = accB[mt][nt][v];
        float s = a / (1.0f + __expf(-a));           // silu
        Hid[(size_t)row * HDIM + col] = f2bf(s * b);
      }
}

// ---------------- plain GEMM: Out = Hs @ Wout (f32 result, overwrite) -----
__global__ __launch_bounds__(256) void gemm_out_kernel(
    const unsigned short* __restrict__ A,   // [S, H] bf16
    const unsigned short* __restrict__ W,   // [H, C] bf16
    float* __restrict__ Out) {              // [S, C] f32
  __shared__ unsigned short sA[128 * STR];
  __shared__ unsigned short sB[64 * STR];
  int tid = threadIdx.x, lane = tid & 31, wave = tid >> 5;
  int wm = wave & 3, wn = wave >> 2;
  int m0 = blockIdx.y * 128, n0 = blockIdx.x * 64;
  v8f acc[2][2] = {};
  for (int k0 = 0; k0 < HDIM; k0 += LDK) {
    stage_a(sA, A, m0, k0, HDIM, tid);
    stage_b(sB, W, k0, n0, CDIM, tid);
    __syncthreads();
#pragma unroll
    for (int kc = 0; kc < 2; ++kc) {
      v16bf a0 = load_a_frag(sA, wm * 32,      kc, lane);
      v16bf a1 = load_a_frag(sA, wm * 32 + 16, kc, lane);
      v16bf b0 = load_b_frag(sB, wn * 32,      kc, lane);
      v16bf b1 = load_b_frag(sB, wn * 32 + 16, kc, lane);
      acc[0][0] = wmma_bf16(a0, b0, acc[0][0]);
      acc[0][1] = wmma_bf16(a0, b1, acc[0][1]);
      acc[1][0] = wmma_bf16(a1, b0, acc[1][0]);
      acc[1][1] = wmma_bf16(a1, b1, acc[1][1]);
    }
    __syncthreads();
  }
  int rofs = (lane < 16) ? 0 : 8;
  int cofs = lane & 15;
#pragma unroll
  for (int mt = 0; mt < 2; ++mt)
#pragma unroll
    for (int nt = 0; nt < 2; ++nt)
#pragma unroll
      for (int v = 0; v < 8; ++v) {
        int row = m0 + wm * 32 + mt * 16 + rofs + v;
        int col = n0 + wn * 32 + nt * 16 + cofs;
        Out[(size_t)row * CDIM + col] = acc[mt][nt][v];
      }
}

// ---------------- routed experts: Out += sum_e wdense[:,e] * (Hr @ W2[e]) -
__global__ __launch_bounds__(256) void gemm_routed_kernel(
    const unsigned short* __restrict__ Hr,   // [S, H] bf16
    const unsigned short* __restrict__ W2,   // [E, H, C] bf16
    const float* __restrict__ Wd,            // [S, E] dispatch weights
    float* __restrict__ Out) {               // [S, C] f32 (accumulate)
  __shared__ unsigned short sA[128 * STR];
  __shared__ unsigned short sB[64 * STR];
  int tid = threadIdx.x, lane = tid & 31, wave = tid >> 5;
  int wm = wave & 3, wn = wave >> 2;
  int m0 = blockIdx.y * 128, n0 = blockIdx.x * 64;
  int rofs = (lane < 16) ? 0 : 8;
  int cofs = lane & 15;
  v8f tot[2][2] = {};
  for (int e = 0; e < NEXP; ++e) {
    const unsigned short* W = W2 + (size_t)e * HDIM * CDIM;
    v8f acc[2][2] = {};
    for (int k0 = 0; k0 < HDIM; k0 += LDK) {
      stage_a(sA, Hr, m0, k0, HDIM, tid);
      stage_b(sB, W, k0, n0, CDIM, tid);
      __syncthreads();
#pragma unroll
      for (int kc = 0; kc < 2; ++kc) {
        v16bf a0 = load_a_frag(sA, wm * 32,      kc, lane);
        v16bf a1 = load_a_frag(sA, wm * 32 + 16, kc, lane);
        v16bf b0 = load_b_frag(sB, wn * 32,      kc, lane);
        v16bf b1 = load_b_frag(sB, wn * 32 + 16, kc, lane);
        acc[0][0] = wmma_bf16(a0, b0, acc[0][0]);
        acc[0][1] = wmma_bf16(a0, b1, acc[0][1]);
        acc[1][0] = wmma_bf16(a1, b0, acc[1][0]);
        acc[1][1] = wmma_bf16(a1, b1, acc[1][1]);
      }
      __syncthreads();
    }
#pragma unroll
    for (int mt = 0; mt < 2; ++mt) {
      int rbase = m0 + wm * 32 + mt * 16 + rofs;
#pragma unroll
      for (int v = 0; v < 8; ++v) {
        float wd = Wd[(size_t)(rbase + v) * NEXP + e];
#pragma unroll
        for (int nt = 0; nt < 2; ++nt)
          tot[mt][nt][v] += wd * acc[mt][nt][v];
      }
    }
  }
#pragma unroll
  for (int mt = 0; mt < 2; ++mt)
#pragma unroll
    for (int nt = 0; nt < 2; ++nt)
#pragma unroll
      for (int v = 0; v < 8; ++v) {
        int row = m0 + wm * 32 + mt * 16 + rofs + v;
        int col = n0 + wn * 32 + nt * 16 + cofs;
        size_t idx = (size_t)row * CDIM + col;
        Out[idx] += tot[mt][nt][v];
      }
}

extern "C" void kernel_launch(void* const* d_in, const int* in_sizes, int n_in,
                              void* d_out, int out_size, void* d_ws, size_t ws_size,
                              hipStream_t stream) {
  const float* x            = (const float*)d_in[0];  // [S, C]
  const float* w_shared_in  = (const float*)d_in[1];  // [C, 2H]
  const float* w_shared_out = (const float*)d_in[2];  // [H, C]
  const float* w1_shared    = (const float*)d_in[3];  // [C, 2H]
  const float* w2           = (const float*)d_in[4];  // [E, H, C]
  const float* w_gate       = (const float*)d_in[5];  // [C, E]
  const float* gate_bias    = (const float*)d_in[6];  // [E]
  float* out = (float*)d_out;

  // workspace layout (bf16 stored as ushort)
  unsigned short* xb      = (unsigned short*)d_ws;                 // S*C
  unsigned short* w_in_b  = xb      + (size_t)S_TOK * CDIM;        // C*2H
  unsigned short* w1_b    = w_in_b  + (size_t)CDIM * 2 * HDIM;     // C*2H
  unsigned short* w_out_b = w1_b    + (size_t)CDIM * 2 * HDIM;     // H*C
  unsigned short* w2_b    = w_out_b + (size_t)HDIM * CDIM;         // E*H*C
  unsigned short* hshared = w2_b    + (size_t)NEXP * HDIM * CDIM;  // S*H
  unsigned short* hrouted = hshared + (size_t)S_TOK * HDIM;        // S*H
  float*          wdense  = (float*)(hrouted + (size_t)S_TOK * HDIM); // S*E

  // 1) cast everything to bf16
  auto cast = [&](const float* s, unsigned short* d, size_t n) {
    int n4 = (int)(n / 4);
    cast_bf16_kernel<<<(n4 + 255) / 256, 256, 0, stream>>>(s, d, n4);
  };
  cast(x,            xb,      (size_t)S_TOK * CDIM);
  cast(w_shared_in,  w_in_b,  (size_t)CDIM * 2 * HDIM);
  cast(w1_shared,    w1_b,    (size_t)CDIM * 2 * HDIM);
  cast(w_shared_out, w_out_b, (size_t)HDIM * CDIM);
  cast(w2,           w2_b,    (size_t)NEXP * HDIM * CDIM);

  // 2) router
  gate_kernel<<<S_TOK / 8, 256, 0, stream>>>(x, w_gate, gate_bias, wdense);

  // 3) SwiGLU hiddens (shared expert + routed shared-w1 hidden)
  dim3 grid(HDIM / 64, S_TOK / 128);
  gemm_swiglu_kernel<<<grid, 256, 0, stream>>>(xb, w_in_b, hshared);
  gemm_swiglu_kernel<<<grid, 256, 0, stream>>>(xb, w1_b,   hrouted);

  // 4) shared expert output (overwrites d_out)
  dim3 grid2(CDIM / 64, S_TOK / 128);
  gemm_out_kernel<<<grid2, 256, 0, stream>>>(hshared, w_out_b, out);

  // 5) routed experts accumulate into d_out
  gemm_routed_kernel<<<grid2, 256, 0, stream>>>(hrouted, w2_b, wdense, out);
}
